// PicoDetHead_30331059044645
// MI455X (gfx1250) — compile-verified, matrix operand order checked by reading, not run on hardware
//
#include <hip/hip_runtime.h>

typedef __attribute__((ext_vector_type(16))) _Float16 v16h;
typedef __attribute__((ext_vector_type(8)))  _Float16 v8h;
typedef __attribute__((ext_vector_type(8)))  float    v8f;

#define BN_EPS 1e-5f

// ---------------------------------------------------------------------------
// NCHW f32 -> pixel-major [P][96] f16, LDS-tiled transpose (coalesced both ways)
// ---------------------------------------------------------------------------
__global__ void __launch_bounds__(256) to_pix_f16(const float* __restrict__ in,
                                                  _Float16* __restrict__ out,
                                                  int HW) {
  __shared__ _Float16 tile[96 * 64];
  const int b = blockIdx.y;
  const int hw0 = blockIdx.x * 64;
  for (int i = threadIdx.x; i < 96 * 64; i += 256) {
    int c = i >> 6, j = i & 63;
    tile[(c << 6) + j] = (_Float16)in[((size_t)b * 96 + c) * HW + hw0 + j];
  }
  __syncthreads();
  for (int i = threadIdx.x; i < 96 * 64; i += 256) {
    int j = i / 96, c = i % 96;
    out[((size_t)(b * HW + hw0 + j)) * 96 + c] = tile[(c << 6) + j];
  }
}

__global__ void fill_zero_f16(_Float16* __restrict__ p, int n) {
  int i = blockIdx.x * blockDim.x + threadIdx.x;
  if (i < n) p[i] = (_Float16)0.f;
}

// ---------------------------------------------------------------------------
// Fold BN into pointwise conv weights and emit WMMA B-fragment layout:
//   wfrag[nt][t][ks][lane][j(16 halves)]
// lane<16 holds N=nt*16+lane, K in {ks*32+0..7, ks*32+16..23}
// lane>=16 holds N=nt*16+lane-16, K in {ks*32+8..15, ks*32+24..31}
// Also writes bias_out[0..Nt*16) (BN shift or pred bias, zero padded).
// ---------------------------------------------------------------------------
__global__ void prep_wfrag(const float* __restrict__ w,       // [Cout][96][taps]
                           const float* __restrict__ gamma, const float* __restrict__ beta,
                           const float* __restrict__ mean,  const float* __restrict__ var,
                           const float* __restrict__ bias_in,
                           _Float16* __restrict__ wfrag, float* __restrict__ bias_out,
                           int Cout, int taps, int Nt) {
  int idx = blockIdx.x * blockDim.x + threadIdx.x;
  if (idx < Nt * 16) {
    float bo = 0.f;
    if (idx < Cout) {
      if (gamma) {
        float s = gamma[idx] * rsqrtf(var[idx] + BN_EPS);
        bo = beta[idx] - mean[idx] * s;
      } else {
        bo = bias_in[idx];
      }
    }
    bias_out[idx] = bo;
  }
  int total = Nt * taps * 1536;
  if (idx >= total) return;
  int j    = idx & 15;
  int lane = (idx >> 4) & 31;
  int ks   = (idx >> 9) % 3;
  int t    = (idx / 1536) % taps;
  int nt   = idx / (1536 * taps);
  int n = nt * 16 + (lane & 15);
  int kbase = (lane < 16) ? 0 : 8;
  int k = ks * 32 + kbase + (j & 7) + ((j >> 3) << 4);
  float v = 0.f;
  if (n < Cout) {
    float s = 1.f;
    if (gamma) s = gamma[n] * rsqrtf(var[n] + BN_EPS);
    v = w[((size_t)n * 96 + k) * taps + t] * s;
  }
  wfrag[idx] = (_Float16)v;
}

// Fold BN scale into depthwise/grouped 3x3 weights (stay fp32, tiny).
__global__ void prep_dw(const float* __restrict__ cw, const float* __restrict__ gamma,
                        const float* __restrict__ beta, const float* __restrict__ mean,
                        const float* __restrict__ var,
                        float* __restrict__ wout, float* __restrict__ bout, int Cc) {
  int idx = blockIdx.x * blockDim.x + threadIdx.x;
  if (idx < Cc) {
    float s = gamma[idx] * rsqrtf(var[idx] + BN_EPS);
    bout[idx] = beta[idx] - mean[idx] * s;
  }
  if (idx >= Cc * 9) return;
  int oc = idx / 9;
  float s = gamma[oc] * rsqrtf(var[oc] + BN_EPS);
  wout[idx] = cw[idx] * s;
}

// ---------------------------------------------------------------------------
// WMMA pointwise/3x3 conv GEMM.
//  - 4 waves/block; each wave computes FOUR 16(M-pixels)x16(N-chan) tiles
//    (4 independent accumulator chains -> XDL pipelining, B-frag reused 4x)
//  - all 4 A-fragments loaded into distinct registers BEFORE the WMMA burst
//    so 8 b128 loads are in flight and waits drain progressively
//  - weights for this N-tile staged in (dynamic) LDS in fragment order
//  - TAPS=9: conv3x3 as 9 shift-accumulated GEMMs; border rows read a zeroed
//    guard row via branchless pointer select (no EXEC churn around WMMA)
//  - all spatial dims are powers of two -> shift/mask addressing only
//  - fused bias (+ ReLU6), f16 or f32 output
// Requires P % 256 == 0 (true for all levels here).
// ---------------------------------------------------------------------------
template <int TAPS, bool RELU6, bool OUTF16>
__global__ void __launch_bounds__(128) gemm_pw(
    const _Float16* __restrict__ A, const _Float16* __restrict__ zrow,
    int P, int wShift, int hwShift,
    const _Float16* __restrict__ Wfrag, const float* __restrict__ bias,
    int Cout, void* __restrict__ Out, int outStride) {
  extern __shared__ _Float16 ldsW[];
  const int nt = blockIdx.y;
  {
    const uint4* src = (const uint4*)(Wfrag + (size_t)nt * TAPS * 1536);
    uint4* dst = (uint4*)ldsW;
#pragma unroll
    for (int i = threadIdx.x; i < TAPS * 192; i += 128) dst[i] = src[i];
  }
  __syncthreads();

  const int lane  = threadIdx.x & 31;
  const int warp  = threadIdx.x >> 5;
  const int r15   = lane & 15;
  const int kbase = (lane < 16) ? 0 : 8;
  const int mt0   = (blockIdx.x * 4 + warp) * 4;  // 4 M-tiles per wave
  if (mt0 * 16 >= P) return;

  const int Ww = 1 << wShift;
  const int Hh = 1 << (hwShift - wShift);

  const _Float16* abase[4];
  int ys[4], xs[4];
#pragma unroll
  for (int s = 0; s < 4; ++s) {
    int row = (mt0 + s) * 16 + r15;
    abase[s] = A + (size_t)row * 96;
    if (TAPS > 1) {
      int hw = row & ((1 << hwShift) - 1);
      ys[s] = hw >> wShift;
      xs[s] = hw & (Ww - 1);
    }
  }

  v8f acc[4];
#pragma unroll
  for (int s = 0; s < 4; ++s)
#pragma unroll
    for (int r = 0; r < 8; ++r) acc[s][r] = 0.f;

#pragma unroll
  for (int t = 0; t < TAPS; ++t) {
    const int dy = (TAPS == 1) ? 0 : (t / 3 - 1);
    const int dx = (TAPS == 1) ? 0 : (t % 3 - 1);
    const _Float16* rp[4];
    if (TAPS > 1) {
      const int off = ((dy << wShift) + dx) * 96;  // shifted row == row + dy*W + dx
#pragma unroll
      for (int s = 0; s < 4; ++s) {
        bool v = ((unsigned)(ys[s] + dy) < (unsigned)Hh) &
                 ((unsigned)(xs[s] + dx) < (unsigned)Ww);
        rp[s] = v ? (abase[s] + off) : zrow;
      }
    } else {
#pragma unroll
      for (int s = 0; s < 4; ++s) rp[s] = abase[s];
    }
#pragma unroll
    for (int ks = 0; ks < 3; ++ks) {
      const v16h bf = *(const v16h*)(ldsW + (((t * 3 + ks) * 32 + lane) << 4));
      union U { v16h v; v8h h[2]; } a[4];
#pragma unroll
      for (int s = 0; s < 4; ++s) {
        a[s].h[0] = *(const v8h*)(rp[s] + ks * 32 + kbase);
        a[s].h[1] = *(const v8h*)(rp[s] + ks * 32 + 16 + kbase);
      }
#pragma unroll
      for (int s = 0; s < 4; ++s) {
        acc[s] = __builtin_amdgcn_wmma_f32_16x16x32_f16(false, a[s].v, false, bf,
                                                        (short)0, acc[s], false, false);
      }
    }
  }

  const int ncol = nt * 16 + r15;
  if (ncol >= Cout) return;
  const float bn = bias[ncol];
#pragma unroll
  for (int s = 0; s < 4; ++s) {
#pragma unroll
    for (int r = 0; r < 8; ++r) {
      float v = acc[s][r] + bn;
      if (RELU6) v = fminf(fmaxf(v, 0.f), 6.f);
      size_t o = (size_t)((mt0 + s) * 16 + kbase + r) * outStride + ncol;
      if (OUTF16) ((_Float16*)Out)[o] = (_Float16)v;
      else        ((float*)Out)[o] = v;
    }
  }
}

// ---------------------------------------------------------------------------
// Ghost "cheap" branch: grouped 3x3 conv (PEROUT outputs per input channel)
// reads cols [0,Cprim), writes cols [Cprim, Cprim+CCHEAP) of the same buffer.
// ---------------------------------------------------------------------------
template <int CCHEAP, int PEROUT>
__global__ void cheap_conv(_Float16* __restrict__ buf, const float* __restrict__ w9,
                           const float* __restrict__ bias, int P, int wShift,
                           int hwShift, int Cprim) {
  int idx = blockIdx.x * blockDim.x + threadIdx.x;
  if (idx >= P * CCHEAP) return;
  int oc = idx % CCHEAP;
  int p  = idx / CCHEAP;
  const int Ww = 1 << wShift;
  const int Hh = 1 << (hwShift - wShift);
  int hw = p & ((1 << hwShift) - 1);
  int y = hw >> wShift, x = hw & (Ww - 1);
  int ic = oc / PEROUT;
  float acc = 0.f;
#pragma unroll
  for (int t = 0; t < 9; ++t) {
    int dy = t / 3 - 1, dx = t % 3 - 1;
    if ((unsigned)(y + dy) < (unsigned)Hh && (unsigned)(x + dx) < (unsigned)Ww)
      acc += w9[oc * 9 + t] * (float)buf[(size_t)(p + (dy << wShift) + dx) * 96 + ic];
  }
  float v = fminf(fmaxf(acc + bias[oc], 0.f), 6.f);
  buf[(size_t)p * 96 + Cprim + oc] = (_Float16)v;
}

// ---------------------------------------------------------------------------
// Decode: DFL softmax expectation -> boxes; sigmoid * logit_scale -> scores.
// ---------------------------------------------------------------------------
__global__ void decode_kernel(const float* __restrict__ cls, const float* __restrict__ reg,
                              const float* __restrict__ logit_scale,
                              float* __restrict__ boxes, float* __restrict__ scores,
                              int P, int wShift, int hwShift, float stridef,
                              int levelOff, int totalA) {
  int p = blockIdx.x * blockDim.x + threadIdx.x;
  if (p >= P) return;
  int b = p >> hwShift;
  int hw = p & ((1 << hwShift) - 1);
  int y = hw >> wShift, x = hw & ((1 << wShift) - 1);
  const float* r = reg + (size_t)p * 36;
  float ltrb[4];
#pragma unroll
  for (int s = 0; s < 4; ++s) {
    float m = r[s * 9];
#pragma unroll
    for (int i = 1; i < 9; ++i) m = fmaxf(m, r[s * 9 + i]);
    float se = 0.f, sw = 0.f;
#pragma unroll
    for (int i = 0; i < 9; ++i) {
      float e = __expf(r[s * 9 + i] - m);
      se += e;
      sw += e * (float)i;
    }
    ltrb[s] = (sw / se) * stridef;
  }
  float cx = ((float)x + 0.5f) * stridef, cy = ((float)y + 0.5f) * stridef;
  size_t a = (size_t)b * totalA + levelOff + hw;
  float* bx = boxes + a * 4;
  bx[0] = cx - ltrb[0];
  bx[1] = cy - ltrb[1];
  bx[2] = cx + ltrb[2];
  bx[3] = cy + ltrb[3];
  float ls = logit_scale[0];
  const float* c = cls + (size_t)p * 80;
  float* sc = scores + a * 80;
  for (int j = 0; j < 80; ++j) sc[j] = ls / (1.f + __expf(-c[j]));
}

// ---------------------------------------------------------------------------
// Host orchestration
// ---------------------------------------------------------------------------
namespace {
struct PW { _Float16* wf; float* bo; int Cout, taps, Nt; };
}

extern "C" void kernel_launch(void* const* d_in, const int* in_sizes, int n_in,
                              void* d_out, int out_size, void* d_ws, size_t ws_size,
                              hipStream_t stream) {
  (void)in_sizes; (void)n_in; (void)out_size; (void)ws_size;

  const int Bsz = 32, TOTAL_A = 5376;
  const int wsh[3] = {6, 5, 4};     // log2(W)
  const int hwsh[3] = {12, 10, 8};  // log2(H*W)
  const int strid[3] = {8, 16, 32};
  const int offA[3] = {0, 4096, 5120};
  const int P0 = Bsz * 64 * 64;  // 131072 (max pixels, level 0)

  // Input index map (jax tree flatten order: sorted dict keys, lists in order)
  // ghost members: cbn{beta,gamma,mean,var}, cw, pbn{beta,gamma,mean,var}, pw
  auto F = [&](int i) { return (const float*)d_in[i]; };
  const int CLS_G[3] = {3, 13, 23};   // cls_conv ghost bases
  const int REG_G[2] = {40, 50};      // reg_conv ghost bases
  const int G_CBN_B = 0, G_CBN_G = 1, G_CBN_M = 2, G_CBN_V = 3, G_CW = 4;
  const int G_PBN_B = 5, G_PBN_G = 6, G_PBN_M = 7, G_PBN_V = 8, G_PW = 9;

  // Workspace bump allocator
  char* ws = (char*)d_ws;
  size_t off = 0;
  auto alloc = [&](size_t bytes) -> void* {
    void* p = ws + off;
    off = (off + bytes + 255) & ~(size_t)255;
    return p;
  };

  _Float16* X  = (_Float16*)alloc((size_t)P0 * 96 * 2);
  _Float16* G1 = (_Float16*)alloc((size_t)P0 * 96 * 2);
  _Float16* G2 = (_Float16*)alloc((size_t)P0 * 96 * 2);
  float* CLS = (float*)alloc((size_t)P0 * 80 * 4);
  float* REG = (float*)alloc((size_t)P0 * 36 * 4);
  _Float16* zrow = (_Float16*)alloc(96 * 2);  // zero guard row for borders

  fill_zero_f16<<<1, 96, 0, stream>>>(zrow, 96);

  PW cls_pw[3], reg_pw[2], clsp[3], regp[3];
  float *dwc_w[3], *dwc_b[3], *dwr_w[2], *dwr_b[2];

  auto do_prep = [&](const float* w, const float* g, const float* be, const float* me,
                     const float* va, const float* bin, PW& pw) {
    pw.wf = (_Float16*)alloc((size_t)pw.Nt * pw.taps * 1536 * 2);
    pw.bo = (float*)alloc((size_t)pw.Nt * 16 * 4);
    int total = pw.Nt * pw.taps * 1536;
    prep_wfrag<<<(total + 255) / 256, 256, 0, stream>>>(w, g, be, me, va, bin, pw.wf,
                                                        pw.bo, pw.Cout, pw.taps, pw.Nt);
  };

  for (int g = 0; g < 3; ++g) {
    int base = CLS_G[g];
    cls_pw[g] = {nullptr, nullptr, 48, (g == 0) ? 9 : 1, 3};
    do_prep(F(base + G_PW), F(base + G_PBN_G), F(base + G_PBN_B), F(base + G_PBN_M),
            F(base + G_PBN_V), nullptr, cls_pw[g]);
    dwc_w[g] = (float*)alloc(48 * 9 * 4);
    dwc_b[g] = (float*)alloc(48 * 4);
    prep_dw<<<(48 * 9 + 255) / 256, 256, 0, stream>>>(
        F(base + G_CW), F(base + G_CBN_G), F(base + G_CBN_B), F(base + G_CBN_M),
        F(base + G_CBN_V), dwc_w[g], dwc_b[g], 48);
  }
  for (int g = 0; g < 2; ++g) {
    int base = REG_G[g];
    reg_pw[g] = {nullptr, nullptr, 32, 1, 2};
    do_prep(F(base + G_PW), F(base + G_PBN_G), F(base + G_PBN_B), F(base + G_PBN_M),
            F(base + G_PBN_V), nullptr, reg_pw[g]);
    dwr_w[g] = (float*)alloc(64 * 9 * 4);
    dwr_b[g] = (float*)alloc(64 * 4);
    prep_dw<<<(64 * 9 + 255) / 256, 256, 0, stream>>>(
        F(base + G_CW), F(base + G_CBN_G), F(base + G_CBN_B), F(base + G_CBN_M),
        F(base + G_CBN_V), dwr_w[g], dwr_b[g], 64);
  }
  for (int i = 0; i < 3; ++i) {
    clsp[i] = {nullptr, nullptr, 80, 1, 5};
    do_prep(F(34 + 2 * i), nullptr, nullptr, nullptr, nullptr, F(33 + 2 * i), clsp[i]);
    regp[i] = {nullptr, nullptr, 36, 1, 3};
    do_prep(F(61 + 2 * i), nullptr, nullptr, nullptr, nullptr, F(60 + 2 * i), regp[i]);
  }

  auto gemm9 = [&](const _Float16* A, int P, int ws_, int hws, const PW& pw,
                   void* Out, int ostride) {
    dim3 grid(P / 256, pw.Nt);
    gemm_pw<9, true, true><<<grid, 128, 9 * 3072, stream>>>(
        A, zrow, P, ws_, hws, pw.wf, pw.bo, pw.Cout, Out, ostride);
  };
  auto gemm1 = [&](const _Float16* A, int P, int ws_, int hws, const PW& pw,
                   void* Out, int ostride, bool relu6) {
    dim3 grid(P / 256, pw.Nt);
    if (relu6)
      gemm_pw<1, true, true><<<grid, 128, 3072, stream>>>(
          A, zrow, P, ws_, hws, pw.wf, pw.bo, pw.Cout, Out, ostride);
    else
      gemm_pw<1, false, false><<<grid, 128, 3072, stream>>>(
          A, zrow, P, ws_, hws, pw.wf, pw.bo, pw.Cout, Out, ostride);
  };

  float* boxes = (float*)d_out;
  float* scores = boxes + (size_t)Bsz * TOTAL_A * 4;
  const float* logit_scale = F(39);

  for (int L = 0; L < 3; ++L) {
    const int HW = 1 << hwsh[L], P = Bsz * HW;
    const int WS = wsh[L], HS = hwsh[L];
    const float* feats = F(L);

    dim3 tgrid(HW / 64, Bsz);
    to_pix_f16<<<tgrid, 256, 0, stream>>>(feats, X, HW);

    // cls branch: Ghost x3
    gemm9(X, P, WS, HS, cls_pw[0], G1, 96);
    cheap_conv<48, 1><<<(P * 48 + 255) / 256, 256, 0, stream>>>(G1, dwc_w[0], dwc_b[0],
                                                                P, WS, HS, 48);
    gemm1(G1, P, WS, HS, cls_pw[1], G2, 96, true);
    cheap_conv<48, 1><<<(P * 48 + 255) / 256, 256, 0, stream>>>(G2, dwc_w[1], dwc_b[1],
                                                                P, WS, HS, 48);
    gemm1(G2, P, WS, HS, cls_pw[2], G1, 96, true);
    cheap_conv<48, 1><<<(P * 48 + 255) / 256, 256, 0, stream>>>(G1, dwc_w[2], dwc_b[2],
                                                                P, WS, HS, 48);
    gemm1(G1, P, WS, HS, clsp[L], CLS, 80, false);

    // reg branch: Ghost x2
    gemm1(X, P, WS, HS, reg_pw[0], G2, 96, true);
    cheap_conv<64, 2><<<(P * 64 + 255) / 256, 256, 0, stream>>>(G2, dwr_w[0], dwr_b[0],
                                                                P, WS, HS, 32);
    gemm1(G2, P, WS, HS, reg_pw[1], G1, 96, true);
    cheap_conv<64, 2><<<(P * 64 + 255) / 256, 256, 0, stream>>>(G1, dwr_w[1], dwr_b[1],
                                                                P, WS, HS, 32);
    gemm1(G1, P, WS, HS, regp[L], REG, 36, false);

    decode_kernel<<<(P + 127) / 128, 128, 0, stream>>>(
        CLS, REG, logit_scale, boxes, scores, P, WS, HS, (float)strid[L], offA[L], TOTAL_A);
  }
}
